// Attention_79439715107080
// MI455X (gfx1250) — matrix operand order repaired and checked by reading, not compiled, not run on hardware
//
#include <hip/hip_runtime.h>

// ---------------------------------------------------------------------------
// GQA attention block for gfx1250 (MI455X), all GEMMs via v_wmma_f32_16x16x32_f16
// ---------------------------------------------------------------------------

#define D_MODEL   4096
#define NUM_HEADS 32
#define NUM_KV    8
#define HEAD_DIM  128
#define GROUP     4
#define SEQ       2048
#define BATCH     2
#define QKV_DIM   6144            // 4096 Q + 1024 K + 1024 V
#define BS        (BATCH * SEQ)   // 4096 tokens

typedef __attribute__((ext_vector_type(16))) _Float16 v16h;
typedef __attribute__((ext_vector_type(8)))  float    v8f;

// Load a 16-half fragment as two 16B LDS reads (ds_load_b128 x2).
static __device__ __forceinline__ v16h frag_ld(const _Float16* p0, const _Float16* p1) {
    v16h r;
    ((uint4*)&r)[0] = *(const uint4*)p0;
    ((uint4*)&r)[1] = *(const uint4*)p1;
    return r;
}

static __device__ __forceinline__ float rmax16(float v) {
    v = fmaxf(v, __shfl_xor(v, 1, 32));
    v = fmaxf(v, __shfl_xor(v, 2, 32));
    v = fmaxf(v, __shfl_xor(v, 4, 32));
    v = fmaxf(v, __shfl_xor(v, 8, 32));
    return v;   // xor masks < 16 never cross the 16-lane half of a wave32
}
static __device__ __forceinline__ float rsum16(float v) {
    v += __shfl_xor(v, 1, 32);
    v += __shfl_xor(v, 2, 32);
    v += __shfl_xor(v, 4, 32);
    v += __shfl_xor(v, 8, 32);
    return v;
}

// ---------------------------------------------------------------------------
// fp32 -> fp16 (vectorized 4-wide)
// ---------------------------------------------------------------------------
__global__ void f32_to_f16_kernel(const float* __restrict__ src,
                                  _Float16* __restrict__ dst, int n4) {
    int i = blockIdx.x * blockDim.x + threadIdx.x;
    if (i >= n4) return;
    float4 v = ((const float4*)src)[i];
    _Float16 h0 = (_Float16)v.x, h1 = (_Float16)v.y,
             h2 = (_Float16)v.z, h3 = (_Float16)v.w;
    _Float16* d = dst + (size_t)i * 4;
    d[0] = h0; d[1] = h1; d[2] = h2; d[3] = h3;
}

// ---------------------------------------------------------------------------
// GEMM: C[M x N] = A[M x K] (f16, row-major) * W[N x K]^T (f16) + bias (f32)
// Workgroup: 256 threads (8 waves, 4x2 grid), tile 128(M) x 128(N), K-step 32.
// Each wave: 32x64 via 2x4 accumulators of v_wmma_f32_16x16x32_f16.
// Double-buffered LDS (40-half padded stride): one barrier per K-step,
// next-tile global loads issued before compute, distance-2 GL2 prefetch.
// ---------------------------------------------------------------------------
__global__ __launch_bounds__(256) void gemm_bias_kernel(
    const _Float16* __restrict__ A, const _Float16* __restrict__ W,
    const float* __restrict__ bias, void* __restrict__ Cout,
    int M, int N, int K, int c_is_f32)
{
    __shared__ __align__(16) _Float16 As[2][128 * 40];
    __shared__ __align__(16) _Float16 Ws[2][128 * 40];

    const int tid  = threadIdx.x;
    const int lane = tid & 31, w = tid >> 5;
    const int wm = w >> 1, wn = w & 1;          // wave grid 4 (M) x 2 (N)
    const int hi = lane >> 4, ln = lane & 15;
    const int m0 = blockIdx.x * 128, n0 = blockIdx.y * 128;

    // Staging map: 128 rows x 32 halves = 512 uint4 per matrix, 2 per thread.
    const int r0 = (tid * 2) >> 2,      cg0 = (tid * 2) & 3;
    const int r1 = (tid * 2 + 1) >> 2,  cg1 = (tid * 2 + 1) & 3;
    const _Float16* Ag0 = A + (size_t)(m0 + r0) * K + cg0 * 8;
    const _Float16* Ag1 = A + (size_t)(m0 + r1) * K + cg1 * 8;
    const _Float16* Wg0 = W + (size_t)(n0 + r0) * K + cg0 * 8;
    const _Float16* Wg1 = W + (size_t)(n0 + r1) * K + cg1 * 8;
    _Float16* As0 = &As[0][0] + r0 * 40 + cg0 * 8;
    _Float16* As1 = &As[0][0] + r1 * 40 + cg1 * 8;
    _Float16* Ws0 = &Ws[0][0] + r0 * 40 + cg0 * 8;
    _Float16* Ws1 = &Ws[0][0] + r1 * 40 + cg1 * 8;
    const int bufStride = 128 * 40;     // halves per buffer

    v8f acc[2][4];
    #pragma unroll
    for (int i = 0; i < 2; ++i)
        #pragma unroll
        for (int j = 0; j < 4; ++j) acc[i][j] = (v8f){};

    // Prologue: stage k0 = 0 into buffer 0.
    {
        uint4 a0 = *(const uint4*)Ag0, a1 = *(const uint4*)Ag1;
        uint4 w0 = *(const uint4*)Wg0, w1 = *(const uint4*)Wg1;
        *(uint4*)As0 = a0; *(uint4*)As1 = a1;
        *(uint4*)Ws0 = w0; *(uint4*)Ws1 = w1;
    }
    __syncthreads();

    int buf = 0;
    for (int k0 = 0; k0 < K; k0 += 32) {
        const bool hasNext = (k0 + 32 < K);

        // Issue next-tile global loads early (latency hidden under WMMAs).
        uint4 a0, a1, w0, w1;
        if (hasNext) {
            a0 = *(const uint4*)(Ag0 + k0 + 32);
            a1 = *(const uint4*)(Ag1 + k0 + 32);
            w0 = *(const uint4*)(Wg0 + k0 + 32);
            w1 = *(const uint4*)(Wg1 + k0 + 32);
            __builtin_prefetch(Ag0 + k0 + 64, 0, 0);   // distance-2 -> GL2
            __builtin_prefetch(Wg0 + k0 + 64, 0, 0);
        }

        // Compute from current buffer.
        const _Float16* Ab = &As[0][0] + buf * bufStride;
        const _Float16* Wb = &Ws[0][0] + buf * bufStride;
        v16h af[2], bf[4];
        #pragma unroll
        for (int i = 0; i < 2; ++i) {
            const _Float16* p = Ab + (wm * 32 + i * 16 + ln) * 40 + hi * 8;
            af[i] = frag_ld(p, p + 16);
        }
        #pragma unroll
        for (int j = 0; j < 4; ++j) {
            const _Float16* p = Wb + (wn * 64 + j * 16 + ln) * 40 + hi * 8;
            bf[j] = frag_ld(p, p + 16);
        }
        #pragma unroll
        for (int i = 0; i < 2; ++i)
            #pragma unroll
            for (int j = 0; j < 4; ++j)
                acc[i][j] = __builtin_amdgcn_wmma_f32_16x16x32_f16(
                    false, af[i], false, bf[j], (short)0, acc[i][j], false, false);

        // Store next tile into the other buffer, one barrier per step.
        if (hasNext) {
            const int nb = (buf ^ 1) * bufStride;
            *(uint4*)(As0 + nb) = a0; *(uint4*)(As1 + nb) = a1;
            *(uint4*)(Ws0 + nb) = w0; *(uint4*)(Ws1 + nb) = w1;
            __syncthreads();
            buf ^= 1;
        }
    }

    // Epilogue: C element (vgpr r, lane) -> row = r + 8*hi, col = ln.
    #pragma unroll
    for (int j = 0; j < 4; ++j) {
        const int col = n0 + wn * 64 + j * 16 + ln;
        const float bj = bias[col];
        #pragma unroll
        for (int i = 0; i < 2; ++i) {
            #pragma unroll
            for (int r = 0; r < 8; ++r) {
                int row = m0 + wm * 32 + i * 16 + r + hi * 8;
                float v = acc[i][j][r] + bj;
                if (c_is_f32) ((float*)Cout)[(size_t)row * N + col] = v;
                else          ((_Float16*)Cout)[(size_t)row * N + col] = (_Float16)v;
            }
        }
    }
}

// ---------------------------------------------------------------------------
// RoPE + split qkv [BS x 6144] -> q[B,H,S,HD], k[B,KV,S,HD] (roped), v[B,KV,S,HD]
// One thread per (even,odd) pair.
// ---------------------------------------------------------------------------
__global__ void rope_split_kernel(const _Float16* __restrict__ qkv,
                                  _Float16* __restrict__ q,
                                  _Float16* __restrict__ k,
                                  _Float16* __restrict__ v)
{
    const int total = BS * (QKV_DIM / 2);
    int tid = blockIdx.x * blockDim.x + threadIdx.x;
    if (tid >= total) return;
    int row = tid / (QKV_DIM / 2);
    int col = (tid - row * (QKV_DIM / 2)) * 2;
    int b = row >> 11, s = row & 2047;

    const _Float16* src = qkv + (size_t)row * QKV_DIM + col;
    float x0 = (float)src[0], x1 = (float)src[1];

    if (col < 5120) {                      // Q or K: apply RoPE
        int seg_col = (col < 4096) ? col : (col - 4096);
        int h = seg_col >> 7, d = seg_col & 127;
        int i = d >> 1;
        float freq = __powf(10000.0f, -(float)(2 * i) * (1.0f / 128.0f));
        float sn, cs;
        __sincosf((float)s * freq, &sn, &cs);
        float er = x0 * cs - x1 * sn;
        float ei = x0 * sn + x1 * cs;
        _Float16* dst;
        if (col < 4096)
            dst = q + (((size_t)(b * NUM_HEADS + h) * SEQ + s) * HEAD_DIM + d);
        else
            dst = k + (((size_t)(b * NUM_KV + h) * SEQ + s) * HEAD_DIM + d);
        dst[0] = (_Float16)er;
        dst[1] = (_Float16)ei;
    } else {                               // V: pass through
        int c3 = col - 5120;
        int kvh = c3 >> 7, d = c3 & 127;
        _Float16* dst = v + (((size_t)(b * NUM_KV + kvh) * SEQ + s) * HEAD_DIM + d);
        dst[0] = src[0];
        dst[1] = src[1];
    }
}

// ---------------------------------------------------------------------------
// Causal flash attention. Grid: (B*H, SEQ/64). Block: 128 threads (4 waves).
// Each wave owns 16 query rows; key blocks of 64.
// Score + PV GEMMs via v_wmma_f32_16x16x32_f16; online softmax in f32.
// ---------------------------------------------------------------------------
__global__ __launch_bounds__(128) void attn_fwd_kernel(
    const _Float16* __restrict__ Q, const _Float16* __restrict__ K,
    const _Float16* __restrict__ V, _Float16* __restrict__ O)
{
    __shared__ __align__(16) _Float16 Qs[64 * 136];   // [qrow][d]
    __shared__ __align__(16) _Float16 Ks[64 * 136];   // [key ][d]
    __shared__ __align__(16) _Float16 Vt[128 * 72];   // [d][key] (transposed)
    __shared__ __align__(16) _Float16 Ps[4][16 * 72]; // per-wave P scratch [m][key]

    const int tid  = threadIdx.x;
    const int lane = tid & 31, w = tid >> 5;
    const int hi = lane >> 4, ln = lane & 15;
    const int bh = blockIdx.x;          // b*32 + h
    const int b  = bh >> 5, h = bh & 31;
    const int kvh = h >> 2;             // GROUP = 4
    const int qBase = blockIdx.y * 64;
    const float smscale = 0.08838834764831845f;   // 1/sqrt(128)

    const _Float16* Qg = Q + ((size_t)bh * SEQ + qBase) * HEAD_DIM;
    const _Float16* Kg = K + ((size_t)(b * NUM_KV + kvh) * SEQ) * HEAD_DIM;
    const _Float16* Vg = V + ((size_t)(b * NUM_KV + kvh) * SEQ) * HEAD_DIM;

    #pragma unroll
    for (int p = 0; p < 8; ++p) {       // stage Q tile 64x128
        int idx = p * 128 + tid;
        int row = idx >> 4, cg = idx & 15;
        *(uint4*)(Qs + row * 136 + cg * 8) =
            *(const uint4*)(Qg + (size_t)row * HEAD_DIM + cg * 8);
    }

    v8f o[8];
    #pragma unroll
    for (int dt = 0; dt < 8; ++dt) o[dt] = (v8f){};
    float mrow[8], lrow[8];
    #pragma unroll
    for (int r = 0; r < 8; ++r) { mrow[r] = -1e30f; lrow[r] = 0.0f; }

    for (int jb = 0; jb <= (int)blockIdx.y; ++jb) {
        const int kBase = jb * 64;
        __syncthreads();                 // prior compute done (also fences Q stage)
        #pragma unroll
        for (int p = 0; p < 8; ++p) {    // stage K tile 64x128
            int idx = p * 128 + tid;
            int row = idx >> 4, cg = idx & 15;
            *(uint4*)(Ks + row * 136 + cg * 8) =
                *(const uint4*)(Kg + (size_t)(kBase + row) * HEAD_DIM + cg * 8);
        }
        #pragma unroll
        for (int p = 0; p < 8; ++p) {    // stage V transposed: Vt[d][key]
            int idx = p * 128 + tid;
            int key = idx & 63, dg = idx >> 6;
            uint4 vv = *(const uint4*)(Vg + (size_t)(kBase + key) * HEAD_DIM + dg * 8);
            const _Float16* hp = (const _Float16*)&vv;
            #pragma unroll
            for (int e = 0; e < 8; ++e) Vt[(dg * 8 + e) * 72 + key] = hp[e];
        }
        __syncthreads();

        // ---- S = Q K^T : 4 key-subtiles x 4 d-steps = 16 WMMAs ----
        v16h qf[4];
        #pragma unroll
        for (int ds = 0; ds < 4; ++ds) {
            const _Float16* qp = Qs + (w * 16 + ln) * 136 + ds * 32 + hi * 8;
            qf[ds] = frag_ld(qp, qp + 16);
        }
        v8f sc[4];
        #pragma unroll
        for (int t = 0; t < 4; ++t) {
            v8f a = {};
            #pragma unroll
            for (int ds = 0; ds < 4; ++ds) {
                const _Float16* kp = Ks + (t * 16 + ln) * 136 + ds * 32 + hi * 8;
                v16h kf = frag_ld(kp, kp + 16);
                a = __builtin_amdgcn_wmma_f32_16x16x32_f16(false, qf[ds], false, kf,
                                                           (short)0, a, false, false);
            }
            sc[t] = a;
        }

        // ---- online softmax (rows live in 16-lane halves) ----
        const bool diag = (jb == (int)blockIdx.y);
        #pragma unroll
        for (int r = 0; r < 8; ++r) {
            int qrow = qBase + w * 16 + r + hi * 8;
            float s0 = sc[0][r] * smscale, s1 = sc[1][r] * smscale;
            float s2 = sc[2][r] * smscale, s3 = sc[3][r] * smscale;
            if (diag) {
                if (kBase +  0 + ln > qrow) s0 = -1e30f;
                if (kBase + 16 + ln > qrow) s1 = -1e30f;
                if (kBase + 32 + ln > qrow) s2 = -1e30f;
                if (kBase + 48 + ln > qrow) s3 = -1e30f;
            }
            float rm = rmax16(fmaxf(fmaxf(s0, s1), fmaxf(s2, s3)));
            float mnew  = fmaxf(mrow[r], rm);
            float alpha = __expf(mrow[r] - mnew);
            float p0 = __expf(s0 - mnew), p1 = __expf(s1 - mnew);
            float p2 = __expf(s2 - mnew), p3 = __expf(s3 - mnew);
            lrow[r] = lrow[r] * alpha + rsum16(p0 + p1 + p2 + p3);
            mrow[r] = mnew;
            #pragma unroll
            for (int dt = 0; dt < 8; ++dt) o[dt][r] *= alpha;   // lane-local rescale
            int prow = r + hi * 8;
            _Float16* pb = &Ps[w][prow * 72 + ln];
            pb[0]  = (_Float16)p0;
            pb[16] = (_Float16)p1;
            pb[32] = (_Float16)p2;
            pb[48] = (_Float16)p3;
        }

        // ---- O += P V : 2 k-steps x 8 d-tiles = 16 WMMAs ----
        #pragma unroll
        for (int ks = 0; ks < 2; ++ks) {
            const _Float16* pp = &Ps[w][ln * 72 + ks * 32 + hi * 8];
            v16h pf = frag_ld(pp, pp + 16);
            #pragma unroll
            for (int dt = 0; dt < 8; ++dt) {
                const _Float16* vp = Vt + (dt * 16 + ln) * 72 + ks * 32 + hi * 8;
                v16h vf = frag_ld(vp, vp + 16);
                o[dt] = __builtin_amdgcn_wmma_f32_16x16x32_f16(false, pf, false, vf,
                                                               (short)0, o[dt], false, false);
            }
        }
    }

    // ---- normalize and store to [b, s, h*128 + d] (f16) ----
    #pragma unroll
    for (int r = 0; r < 8; ++r) {
        float inv = 1.0f / lrow[r];
        int qrow = qBase + w * 16 + r + hi * 8;
        size_t rowoff = ((size_t)b * SEQ + qrow) * D_MODEL + h * HEAD_DIM;
        #pragma unroll
        for (int dt = 0; dt < 8; ++dt)
            O[rowoff + dt * 16 + ln] = (_Float16)(o[dt][r] * inv);
    }
}

// ---------------------------------------------------------------------------
extern "C" void kernel_launch(void* const* d_in, const int* in_sizes, int n_in,
                              void* d_out, int out_size, void* d_ws, size_t ws_size,
                              hipStream_t stream)
{
    (void)in_sizes; (void)n_in; (void)out_size; (void)ws_size;
    const float* x     = (const float*)d_in[0];
    const float* w_qkv = (const float*)d_in[1];
    const float* b_qkv = (const float*)d_in[2];
    const float* w_o   = (const float*)d_in[3];
    const float* b_o   = (const float*)d_in[4];

    char* ws = (char*)d_ws;
    size_t off = 0;
    _Float16* xh    = (_Float16*)(ws + off); off += (size_t)BS * D_MODEL * 2;       // 32 MiB
    _Float16* wqkvh = (_Float16*)(ws + off); off += (size_t)QKV_DIM * D_MODEL * 2;  // 48 MiB
    _Float16* woh   = (_Float16*)(ws + off); off += (size_t)D_MODEL * D_MODEL * 2;  // 32 MiB
    _Float16* qkvh  = (_Float16*)(ws + off); off += (size_t)BS * QKV_DIM * 2;       // 48 MiB
    _Float16* qh    = (_Float16*)(ws + off); off += (size_t)BATCH * NUM_HEADS * SEQ * HEAD_DIM * 2; // 32 MiB
    _Float16* kh    = (_Float16*)(ws + off); off += (size_t)BATCH * NUM_KV * SEQ * HEAD_DIM * 2;    // 8 MiB
    _Float16* vh    = (_Float16*)(ws + off); off += (size_t)BATCH * NUM_KV * SEQ * HEAD_DIM * 2;    // 8 MiB
    _Float16* attn  = xh;   // xh is dead after the QKV GEMM; reuse it           // total ~208 MiB

    // 1) fp32 -> fp16 conversions
    {
        int n4 = (BS * D_MODEL) / 4;
        f32_to_f16_kernel<<<(n4 + 255) / 256, 256, 0, stream>>>(x, xh, n4);
        n4 = (QKV_DIM * D_MODEL) / 4;
        f32_to_f16_kernel<<<(n4 + 255) / 256, 256, 0, stream>>>(w_qkv, wqkvh, n4);
        n4 = (D_MODEL * D_MODEL) / 4;
        f32_to_f16_kernel<<<(n4 + 255) / 256, 256, 0, stream>>>(w_o, woh, n4);
    }

    // 2) QKV projection: [4096 x 4096] @ [6144 x 4096]^T + b_qkv -> f16
    gemm_bias_kernel<<<dim3(BS / 128, QKV_DIM / 128), 256, 0, stream>>>(
        xh, wqkvh, b_qkv, (void*)qkvh, BS, QKV_DIM, D_MODEL, /*c_is_f32=*/0);

    // 3) RoPE + split into head-major q/k/v
    {
        int total = BS * (QKV_DIM / 2);
        rope_split_kernel<<<(total + 255) / 256, 256, 0, stream>>>(qkvh, qh, kh, vh);
    }

    // 4) Causal flash attention -> attn [BS x 4096] f16
    attn_fwd_kernel<<<dim3(BATCH * NUM_HEADS, SEQ / 64), 128, 0, stream>>>(
        qh, kh, vh, attn);

    // 5) Output projection: attn @ w_o^T + b_o -> d_out (f32)
    gemm_bias_kernel<<<dim3(BS / 128, D_MODEL / 128), 256, 0, stream>>>(
        attn, woh, b_o, d_out, BS, D_MODEL, D_MODEL, /*c_is_f32=*/1);
}
